// DeepseekV3DecoderLayer_66486093742696
// MI455X (gfx1250) — compile-verified, matrix operand order checked by reading, not run on hardware
//
#include <hip/hip_runtime.h>

// ---------------------------------------------------------------------------
// DeepseekV3 decoder layer for MI455X (gfx1250, wave32, WMMA, async-LDS).
//  * All GEMMs: v_wmma_f32_16x16x32_bf16, bf16 operands staged in LDS by
//    global_load_async_to_lds_b128 (ASYNCcnt), double-buffered.
//  * Weights pre-converted to bf16 and pre-transposed to [N][K] so both
//    GEMM operands are row-major [row][K] pure copies (no transpose VALU).
//  * Activations produced directly in bf16 by rmsnorm / rope / silu / attn.
//  * All staging loops are uniform (compile-time trip counts) -> no exec
//    masking; attention Q/K tiles are flat contiguous copies (no div/mod).
// ---------------------------------------------------------------------------

#define S_LEN  2048
#define HID    2048
#define NHEAD  16
#define NOPE_D 128
#define ROPE_D 64
#define VHD_D  128
#define QHD_D  192          // NOPE + ROPE
#define QLR_D  1536
#define KVLR_D 512
#define FF_D   8192
#define EPSF   1e-6f
#define SCALEF 0.07216878364870323f   // 192^-0.5

typedef __bf16 bf16;
typedef __attribute__((ext_vector_type(16))) __bf16 v16bf;
typedef __attribute__((ext_vector_type(8)))  float  v8f;

__device__ __forceinline__ bf16 f2bf(float f) {
  union { float f; unsigned u; } x; x.f = f;
  unsigned r = (x.u + 0x7FFFu + ((x.u >> 16) & 1u)) >> 16;
  union { unsigned short s; bf16 b; } y; y.s = (unsigned short)r;
  return y.b;
}
__device__ __forceinline__ float bf2f(bf16 b) {
  union { unsigned short s; bf16 b; } y; y.b = b;
  union { unsigned u; float f; } x; x.u = ((unsigned)y.s) << 16;
  return x.f;
}

__device__ __forceinline__ v8f wmma_bf16(v16bf a, v16bf b, v8f c) {
  return __builtin_amdgcn_wmma_f32_16x16x32_bf16(
      false, a, false, b, (short)0, c, false, false);
}

// Async 16-byte global -> LDS copy (per-lane), tracked by ASYNCcnt.
// LDS address = low 32 bits of the generic pointer (CDNA5 maps the shared
// aperture entirely in addr[63:32]; addr[31:0] is the LDS byte offset).
__device__ __forceinline__ void async_cp16(void* lds, const void* g) {
  unsigned l = (unsigned)(size_t)lds;
  asm volatile("global_load_async_to_lds_b128 %0, %1, off"
               :: "v"(l), "v"(g) : "memory");
}
__device__ __forceinline__ void async_wait0() {
  asm volatile("s_wait_asynccnt 0" ::: "memory");
}

// Load one 16x32 bf16 WMMA operand fragment from LDS; caller passes
// &smem[tileRow0 + (lane & 15)][kk].  Lanes 0-15: K 0-7/16-23, lanes 16-31:
// K 8-15/24-31, bf16 pairs per VGPR -> eight u32 reads (2x ds_load_b128).
__device__ __forceinline__ v16bf frag_ld(const bf16* rowbase, int lane) {
  union { unsigned u[8]; v16bf v; } r;
  const unsigned* p = (const unsigned*)rowbase;
  const int kb = (lane & 16) ? 4 : 0;               // u32 units
#pragma unroll
  for (int i = 0; i < 8; ++i)
    r.u[i] = p[((i & 4) ? 8 : 0) + kb + (i & 3)];
  return r.v;
}

// ---------------------------------------------------------------------------
// Weight prep: Wt[n][k] = bf16(W[k][n]).  32x32 LDS tile transpose.
// ---------------------------------------------------------------------------
__global__ __launch_bounds__(256) void convert_transpose_kernel(
    const float* __restrict__ W, bf16* __restrict__ Wt, int K, int N) {
  __shared__ float t[32][33];
  const int kb = blockIdx.y * 32, nb = blockIdx.x * 32;
  const int tx = threadIdx.x & 31, ty = threadIdx.x >> 5;   // 32 x 8
#pragma unroll
  for (int j = 0; j < 4; ++j)
    t[ty + j * 8][tx] = W[(size_t)(kb + ty + j * 8) * N + nb + tx];
  __syncthreads();
#pragma unroll
  for (int j = 0; j < 4; ++j)
    Wt[(size_t)(nb + ty + j * 8) * K + kb + tx] = f2bf(t[tx][ty + j * 8]);
}

// ---------------------------------------------------------------------------
// GEMM: C[M,N] = A[M,K] @ B[K,N] (+ addend), A bf16 [M][K], Bt bf16 [N][K].
// Block tile 128xBN, K-step 32.  8 waves = 4(m) x 2(n); each wave owns
// 32 rows x BN/2 cols -> 2 A-frags x (BN/32) B-frags -> 8 (BN=128) WMMAs
// per K-step.  Tiles staged by async LDS copies, double-buffered.
// ---------------------------------------------------------------------------
template <int BN, bool FUSE_ADD>
__global__ __launch_bounds__(256) void gemm_bf16(
    const bf16* __restrict__ A, const bf16* __restrict__ Bt,
    const float* __restrict__ addend, float* __restrict__ C,
    int M, int N, int K) {
  constexpr int TNW = BN / 32;              // 16-wide tiles per wave
  constexpr int BTRIP = (BN * 4) / 256;     // B-tile b128 copies per thread
  __shared__ __align__(16) bf16 As[2][128][32];
  __shared__ __align__(16) bf16 Bs[2][BN][32];
  const int tid = threadIdx.x, wave = tid >> 5, lane = tid & 31;
  const int m0 = blockIdx.y * 128, n0 = blockIdx.x * BN;
  const int row0 = (wave >> 1) * 32;        // wave M origin within block
  const int col0 = (wave & 1) * (BN / 2);   // wave N origin within block

  v8f acc[2][TNW];
  v8f zero = {};
#pragma unroll
  for (int a = 0; a < 2; ++a)
#pragma unroll
    for (int b = 0; b < TNW; ++b) acc[a][b] = zero;

  auto stage = [&](int buf, int k0) {
#pragma unroll
    for (int j = 0; j < 2; ++j) {                       // A: 8KB, 2/thread
      int i = tid + j * 256;
      int r = i >> 2, c = i & 3;
      async_cp16(&As[buf][r][c * 8], A + (size_t)(m0 + r) * K + k0 + c * 8);
    }
#pragma unroll
    for (int j = 0; j < BTRIP; ++j) {                   // B: BN*64B
      int i = tid + j * 256;
      int r = i >> 2, c = i & 3;
      async_cp16(&Bs[buf][r][c * 8], Bt + (size_t)(n0 + r) * K + k0 + c * 8);
    }
  };

  stage(0, 0);
  async_wait0();
  __syncthreads();

  int cur = 0;
  for (int k0 = 0; k0 < K; k0 += 32) {
    if (k0 + 32 < K) stage(cur ^ 1, k0 + 32);   // overlap with WMMAs below
    v16bf af[2], bfr[TNW];
#pragma unroll
    for (int a = 0; a < 2; ++a)
      af[a] = frag_ld(&As[cur][row0 + a * 16 + (lane & 15)][0], lane);
#pragma unroll
    for (int b = 0; b < TNW; ++b)
      bfr[b] = frag_ld(&Bs[cur][col0 + b * 16 + (lane & 15)][0], lane);
#pragma unroll
    for (int a = 0; a < 2; ++a)
#pragma unroll
      for (int b = 0; b < TNW; ++b)
        acc[a][b] = wmma_bf16(af[a], bfr[b], acc[a][b]);
    async_wait0();
    __syncthreads();
    cur ^= 1;
  }

  // C/D layout: VGPR r, lane l -> M = r + (l>=16 ? 8 : 0), N = l & 15
  const int mm = (lane >> 4) * 8, nn = lane & 15;
#pragma unroll
  for (int a = 0; a < 2; ++a)
#pragma unroll
    for (int b = 0; b < TNW; ++b)
#pragma unroll
      for (int r = 0; r < 8; ++r) {
        size_t idx = (size_t)(m0 + row0 + a * 16 + mm + r) * N
                   + n0 + col0 + b * 16 + nn;
        float v = acc[a][b][r];
        if (FUSE_ADD) v += addend[idx];
        C[idx] = v;
      }
}

// ---------------------------------------------------------------------------
// RMSNorm, fp32 in -> bf16 out (feeds GEMM A operands).
// ---------------------------------------------------------------------------
__global__ __launch_bounds__(256) void rmsnorm_kernel(
    const float* __restrict__ x, const float* __restrict__ w,
    bf16* __restrict__ y, int dim, int in_stride, int out_stride) {
  const int row = blockIdx.x;
  const float* xr = x + (size_t)row * in_stride;
  bf16* yr = y + (size_t)row * out_stride;
  float ss = 0.f;
  for (int i = threadIdx.x; i < dim; i += 256) { float v = xr[i]; ss += v * v; }
#pragma unroll
  for (int o = 16; o; o >>= 1) ss += __shfl_xor(ss, o, 32);
  __shared__ float red[8];
  if ((threadIdx.x & 31) == 0) red[threadIdx.x >> 5] = ss;
  __syncthreads();
  if (threadIdx.x == 0) {
    float t = 0.f;
    for (int i = 0; i < 8; ++i) t += red[i];
    red[0] = rsqrtf(t / (float)dim + EPSF);
  }
  __syncthreads();
  const float sc = red[0];
  for (int i = threadIdx.x; i < dim; i += 256) yr[i] = f2bf(xr[i] * sc * w[i]);
}

// ---------------------------------------------------------------------------
// RoPE + head-major repack, fp32 -> bf16.  Two heads per pass so all 256
// threads stay active (tid>>7 selects head parity, tid&127 the channel).
//   qs [NH][S][192]  (softmax scale folded in)
//   ks [NH][S][192]
//   vt [NH][128][S]  (V pre-transposed so attention V tiles are raw copies)
// Reference permutation: y[:32]=x[0::2], y[32:]=x[1::2]; out = y*c + rot(y)*s.
// ---------------------------------------------------------------------------
__global__ __launch_bounds__(256) void rope_build_kernel(
    const float* __restrict__ qbuf,    // [S][NH*QHD]
    const float* __restrict__ kvbuf,   // [S][NH*(NOPE+VHD)]
    const float* __restrict__ ckv,     // [S][KVLR+ROPE]
    const float* __restrict__ cosb, const float* __restrict__ sinb,  // [S][64]
    const int* __restrict__ pos,
    bf16* __restrict__ qs, bf16* __restrict__ ks, bf16* __restrict__ vt) {
  const int s = blockIdx.x;
  const int p = pos[s];
  const float* cr = cosb + (size_t)p * ROPE_D;
  const float* sr = sinb + (size_t)p * ROPE_D;
  __shared__ float kpe[ROPE_D];
  if (threadIdx.x < ROPE_D / 2) {
    int i = threadIdx.x;
    float x0 = ckv[(size_t)s * (KVLR_D + ROPE_D) + KVLR_D + 2 * i];
    float x1 = ckv[(size_t)s * (KVLR_D + ROPE_D) + KVLR_D + 2 * i + 1];
    kpe[i]              = x0 * cr[i] - x1 * sr[i];
    kpe[i + ROPE_D / 2] = x1 * cr[i + ROPE_D / 2] + x0 * sr[i + ROPE_D / 2];
  }
  __syncthreads();
  const int hh = threadIdx.x >> 7;       // 0/1: which of the head pair
  const int t  = threadIdx.x & 127;      // channel 0..127
  for (int h = 0; h < NHEAD; h += 2) {
    const int head = h + hh;
    const float* qrow  = qbuf  + (size_t)s * NHEAD * QHD_D + head * QHD_D;
    const float* kvrow = kvbuf + (size_t)s * NHEAD * (NOPE_D + VHD_D)
                               + head * (NOPE_D + VHD_D);
    bf16* qo = qs + ((size_t)head * S_LEN + s) * QHD_D;
    bf16* ko = ks + ((size_t)head * S_LEN + s) * QHD_D;
    qo[t] = f2bf(qrow[t] * SCALEF);
    ko[t] = f2bf(kvrow[t]);
    vt[((size_t)head * VHD_D + t) * S_LEN + s] = f2bf(kvrow[NOPE_D + t]);
    if (t < ROPE_D / 2) {
      float x0 = qrow[NOPE_D + 2 * t], x1 = qrow[NOPE_D + 2 * t + 1];
      qo[NOPE_D + t] = f2bf((x0 * cr[t] - x1 * sr[t]) * SCALEF);
      qo[NOPE_D + ROPE_D / 2 + t] =
          f2bf((x1 * cr[t + ROPE_D / 2] + x0 * sr[t + ROPE_D / 2]) * SCALEF);
      ko[NOPE_D + t]              = f2bf(kpe[t]);
      ko[NOPE_D + ROPE_D / 2 + t] = f2bf(kpe[t + ROPE_D / 2]);
    }
  }
}

// ---------------------------------------------------------------------------
// Causal flash attention per (head, 64-query tile), 32-key steps.
// Q/K tiles are contiguous in global -> flat async copies (no div/mod).
// QK^T and PV on WMMA.  LDS: Q 24K + K 12K + Vt 8K + SP 4K + state < 64KB.
// ---------------------------------------------------------------------------
__global__ __launch_bounds__(256) void mla_attention_kernel(
    const bf16* __restrict__ qs, const bf16* __restrict__ ks,
    const bf16* __restrict__ vt, bf16* __restrict__ out) {
  const int qb = blockIdx.x, h = blockIdx.y;
  const int q0 = qb * 64;
  const int tid = threadIdx.x, wave = tid >> 5, lane = tid & 31;

  __shared__ __align__(16) bf16 Qs[64][QHD_D];
  __shared__ __align__(16) bf16 Ks[32][QHD_D];
  __shared__ __align__(16) bf16 Vt[VHD_D][32];
  __shared__ __align__(16) bf16 SP[64][32];
  __shared__ float mrow[64], lrow[64], arow[64];

  // Q tile: 64 rows x 192 = contiguous 24KB region -> flat copy, 6/thread
  const bf16* Qg = qs + ((size_t)h * S_LEN + q0) * QHD_D;
#pragma unroll
  for (int j = 0; j < 6; ++j) {
    int i = tid + j * 256;
    async_cp16((bf16*)Qs + (size_t)i * 8, Qg + (size_t)i * 8);
  }
  if (tid < 64) { mrow[tid] = -1e30f; lrow[tid] = 0.f; }

  const int tmO = wave >> 1, tnO = (wave & 1) * 4;  // O: 16 rows x 64 cols
  v8f acc[4];
  v8f zero = {};
#pragma unroll
  for (int t = 0; t < 4; ++t) acc[t] = zero;
  __syncthreads();

  const int nkb = 2 * qb + 2;   // 32-key blocks covering keys <= q0+63
  for (int kb = 0; kb < nkb; ++kb) {
    const int k0 = kb * 32;
    const bf16* Kg = ks + ((size_t)h * S_LEN + k0) * QHD_D;  // contiguous 12KB
    const bf16* Vg = vt + (size_t)h * VHD_D * S_LEN + k0;
#pragma unroll
    for (int j = 0; j < 3; ++j) {                 // K tile: flat, 3/thread
      int i = tid + j * 256;
      async_cp16((bf16*)Ks + (size_t)i * 8, Kg + (size_t)i * 8);
    }
    {                                             // V tile: 128 rows x 64B
      int r = tid >> 1, c = tid & 1;
      async_cp16(&Vt[r][c * 16], Vg + (size_t)r * S_LEN + c * 16);
    }
    async_wait0();
    __syncthreads();

    // ---- S = Q @ K^T: 8 tiles (64x32), one per wave ----
    {
      const int tm = wave >> 1, tn = wave & 1;
      v8f sacc = zero;
#pragma unroll
      for (int kk = 0; kk < QHD_D; kk += 32) {
        v16bf a = frag_ld(&Qs[tm * 16 + (lane & 15)][kk], lane);
        v16bf b = frag_ld(&Ks[tn * 16 + (lane & 15)][kk], lane);
        sacc = wmma_bf16(a, b, sacc);
      }
      const int mm = (lane >> 4) * 8, nn = lane & 15;
#pragma unroll
      for (int r = 0; r < 8; ++r) {
        int qi = q0 + tm * 16 + mm + r;
        int ki = k0 + tn * 16 + nn;
        SP[tm * 16 + mm + r][tn * 16 + nn] =
            f2bf((ki <= qi) ? sacc[r] : -1e30f);   // causal mask fused
      }
    }
    __syncthreads();

    // ---- online softmax: 4 threads/row, 8 cols each ----
    {
      const int row = tid >> 2, seg = tid & 3;
      float mx = -1e30f;
#pragma unroll
      for (int c = 0; c < 8; ++c) mx = fmaxf(mx, bf2f(SP[row][seg * 8 + c]));
      mx = fmaxf(mx, __shfl_xor(mx, 1, 32));
      mx = fmaxf(mx, __shfl_xor(mx, 2, 32));
      float m_old = mrow[row];
      float m_new = fmaxf(m_old, mx);
      float ssum = 0.f;
#pragma unroll
      for (int c = 0; c < 8; ++c) {
        float pv = __expf(bf2f(SP[row][seg * 8 + c]) - m_new);
        SP[row][seg * 8 + c] = f2bf(pv);           // S -> P in place
        ssum += pv;
      }
      ssum += __shfl_xor(ssum, 1, 32);
      ssum += __shfl_xor(ssum, 2, 32);
      __syncthreads();                              // all m_old reads done
      if (seg == 0) {
        float alpha = __expf(m_old - m_new);
        arow[row] = alpha;
        lrow[row] = lrow[row] * alpha + ssum;
        mrow[row] = m_new;
      }
    }
    __syncthreads();

    // ---- O = O*alpha + P @ V ----
    {
      const int mm = (lane >> 4) * 8;
      float al[8];
#pragma unroll
      for (int r = 0; r < 8; ++r) al[r] = arow[tmO * 16 + mm + r];
#pragma unroll
      for (int t = 0; t < 4; ++t)
#pragma unroll
        for (int r = 0; r < 8; ++r) acc[t][r] *= al[r];
      v16bf a = frag_ld(&SP[tmO * 16 + (lane & 15)][0], lane);
#pragma unroll
      for (int t = 0; t < 4; ++t) {
        v16bf b = frag_ld(&Vt[(tnO + t) * 16 + (lane & 15)][0], lane);
        acc[t] = wmma_bf16(a, b, acc[t]);
      }
    }
    __syncthreads();
  }

  // normalize and write bf16 out[q][h*VHD + d]  ((B,S,NH*VHD) layout)
  const int mm = (lane >> 4) * 8, nn = lane & 15;
  bf16* Og = out + (size_t)q0 * (NHEAD * VHD_D) + h * VHD_D;
#pragma unroll
  for (int t = 0; t < 4; ++t)
#pragma unroll
    for (int r = 0; r < 8; ++r) {
      int row = tmO * 16 + mm + r;
      int col = (tnO + t) * 16 + nn;
      Og[(size_t)row * (NHEAD * VHD_D) + col] = f2bf(acc[t][r] / lrow[row]);
    }
}

// ---------------------------------------------------------------------------
__global__ void silu_mul_kernel(const float* __restrict__ g,
                                const float* __restrict__ u,
                                bf16* __restrict__ act, size_t n) {
  size_t i = (size_t)blockIdx.x * blockDim.x + threadIdx.x;
  if (i < n) {
    float x = g[i];
    act[i] = f2bf(x / (1.f + __expf(-x)) * u[i]);
  }
}

// ---------------------------------------------------------------------------
extern "C" void kernel_launch(void* const* d_in, const int* in_sizes, int n_in,
                              void* d_out, int out_size, void* d_ws,
                              size_t ws_size, hipStream_t stream) {
  const float* hidden_states = (const float*)d_in[0];
  const float* sinb    = (const float*)d_in[1];
  const float* cosb    = (const float*)d_in[2];
  const float* wq_a    = (const float*)d_in[3];
  const float* q_a_ln  = (const float*)d_in[4];
  const float* wq_b    = (const float*)d_in[5];
  const float* wkv_a   = (const float*)d_in[6];
  const float* kv_a_ln = (const float*)d_in[7];
  const float* wkv_b   = (const float*)d_in[8];
  const float* wo      = (const float*)d_in[9];
  const float* in_ln   = (const float*)d_in[10];
  const float* post_ln = (const float*)d_in[11];
  const float* w_gate  = (const float*)d_in[12];
  const float* w_up    = (const float*)d_in[13];
  const float* w_down  = (const float*)d_in[14];
  const int*   pos     = (const int*)d_in[15];
  float* outp = (float*)d_out;

  char* base = (char*)d_ws;
  size_t off = 0;
  auto allocB = [&](size_t bytes) {
    void* p = base + off;
    off = (off + bytes + 255) & ~(size_t)255;
    return p;
  };
  auto allocF  = [&](size_t n) { return (float*)allocB(n * 4); };
  auto allocH  = [&](size_t n) { return (bf16*)allocB(n * 2); };

  // bf16-transposed weights [N][K]
  bf16* wq_aT   = allocH((size_t)QLR_D * HID);
  bf16* wq_bT   = allocH((size_t)NHEAD * QHD_D * QLR_D);
  bf16* wkv_aT  = allocH((size_t)(KVLR_D + ROPE_D) * HID);
  bf16* wkv_bT  = allocH((size_t)NHEAD * (NOPE_D + VHD_D) * KVLR_D);
  bf16* woT     = allocH((size_t)HID * NHEAD * VHD_D);
  bf16* w_gateT = allocH((size_t)FF_D * HID);
  bf16* w_upT   = allocH((size_t)FF_D * HID);
  bf16* w_downT = allocH((size_t)HID * FF_D);
  // activations
  bf16*  xnorm  = allocH((size_t)S_LEN * HID);               // also y
  float* qlat   = allocF((size_t)S_LEN * QLR_D);
  bf16*  qlatn  = allocH((size_t)S_LEN * QLR_D);
  float* qbuf   = allocF((size_t)S_LEN * NHEAD * QHD_D);
  float* ckv    = allocF((size_t)S_LEN * (KVLR_D + ROPE_D));
  bf16*  kvn    = allocH((size_t)S_LEN * KVLR_D);
  float* kvbuf  = allocF((size_t)S_LEN * NHEAD * (NOPE_D + VHD_D));
  bf16*  qsb    = allocH((size_t)NHEAD * S_LEN * QHD_D);
  bf16*  ksb    = allocH((size_t)NHEAD * S_LEN * QHD_D);
  bf16*  vtb    = allocH((size_t)NHEAD * VHD_D * S_LEN);
  bf16*  attno  = allocH((size_t)S_LEN * NHEAD * VHD_D);
  float* hidden = allocF((size_t)S_LEN * HID);
  float* gate   = allocF((size_t)S_LEN * FF_D);
  float* up     = allocF((size_t)S_LEN * FF_D);
  bf16*  act    = (bf16*)kvbuf;   // dead after rope; 33.5MB >= S*FF bf16

  const dim3 blk(256);
  auto conv = [&](const float* W, bf16* Wt, int K, int N) {
    convert_transpose_kernel<<<dim3(N / 32, K / 32), blk, 0, stream>>>(
        W, Wt, K, N);
  };
  auto gemm = [&](const bf16* A, const bf16* Bt, const float* add, float* C,
                  int M, int N, int K) {
    if (N % 128 == 0) {
      if (add)
        gemm_bf16<128, true><<<dim3(N / 128, M / 128), blk, 0, stream>>>(
            A, Bt, add, C, M, N, K);
      else
        gemm_bf16<128, false><<<dim3(N / 128, M / 128), blk, 0, stream>>>(
            A, Bt, add, C, M, N, K);
    } else {
      gemm_bf16<64, false><<<dim3(N / 64, M / 128), blk, 0, stream>>>(
          A, Bt, add, C, M, N, K);
    }
  };

  // 0) one-time weight conversion + transpose to bf16 [N][K]
  conv(wq_a,   wq_aT,   HID,    QLR_D);
  conv(wq_b,   wq_bT,   QLR_D,  NHEAD * QHD_D);
  conv(wkv_a,  wkv_aT,  HID,    KVLR_D + ROPE_D);
  conv(wkv_b,  wkv_bT,  KVLR_D, NHEAD * (NOPE_D + VHD_D));
  conv(wo,     woT,     NHEAD * VHD_D, HID);
  conv(w_gate, w_gateT, HID,    FF_D);
  conv(w_up,   w_upT,   HID,    FF_D);
  conv(w_down, w_downT, FF_D,   HID);

  // 1) input RMSNorm (fp32 -> bf16)
  rmsnorm_kernel<<<S_LEN, blk, 0, stream>>>(hidden_states, in_ln, xnorm,
                                            HID, HID, HID);
  // 2) q / kv low-rank projections
  gemm(xnorm, wq_aT,  nullptr, qlat, S_LEN, QLR_D, HID);
  gemm(xnorm, wkv_aT, nullptr, ckv,  S_LEN, KVLR_D + ROPE_D, HID);
  // 3) q_a layernorm + up-projection
  rmsnorm_kernel<<<S_LEN, blk, 0, stream>>>(qlat, q_a_ln, qlatn,
                                            QLR_D, QLR_D, QLR_D);
  gemm(qlatn, wq_bT, nullptr, qbuf, S_LEN, NHEAD * QHD_D, QLR_D);
  // 4) kv_a layernorm (first 512 of 576 cols) + up-projection
  rmsnorm_kernel<<<S_LEN, blk, 0, stream>>>(ckv, kv_a_ln, kvn,
                                            KVLR_D, KVLR_D + ROPE_D, KVLR_D);
  gemm(kvn, wkv_bT, nullptr, kvbuf, S_LEN, NHEAD * (NOPE_D + VHD_D), KVLR_D);
  // 5) RoPE + bf16 head-major repack (softmax scale folded into q)
  rope_build_kernel<<<S_LEN, blk, 0, stream>>>(qbuf, kvbuf, ckv, cosb, sinb,
                                               pos, qsb, ksb, vtb);
  // 6) causal flash attention (bf16 in, bf16 out)
  mla_attention_kernel<<<dim3(S_LEN / 64, NHEAD), blk, 0, stream>>>(
      qsb, ksb, vtb, attno);
  // 7) output projection, residual fused into epilogue
  gemm(attno, woT, hidden_states, hidden, S_LEN, HID, NHEAD * VHD_D);
  // 8) post RMSNorm + MLP (residual fused into down-proj epilogue)
  rmsnorm_kernel<<<S_LEN, blk, 0, stream>>>(hidden, post_ln, xnorm,
                                            HID, HID, HID);
  gemm(xnorm, w_gateT, nullptr, gate, S_LEN, FF_D, HID);
  gemm(xnorm, w_upT,   nullptr, up,   S_LEN, FF_D, HID);
  silu_mul_kernel<<<(int)(((size_t)S_LEN * FF_D + 255) / 256), blk, 0, stream>>>(
      gate, up, act, (size_t)S_LEN * FF_D);
  gemm(act, w_downT, hidden, outp, S_LEN, HID, FF_D);

  (void)in_sizes; (void)n_in; (void)out_size; (void)ws_size;
}
